// RNNAgentWithPerception_57647051046981
// MI455X (gfx1250) — matrix operand order repaired and checked by reading, not compiled, not run on hardware
//
#include <hip/hip_runtime.h>

// ---------------------------------------------------------------- types
typedef unsigned short u16;
typedef __attribute__((ext_vector_type(16))) __bf16 v16bf;
typedef __attribute__((ext_vector_type(8)))  float  v8f;
typedef __attribute__((ext_vector_type(8)))  u16    v8us;
typedef __attribute__((ext_vector_type(16))) u16    v16us;
typedef __attribute__((address_space(3)))    u16    lds_u16;

#define MBSZ 8192
#define INSZ 128
#define GHSZ 512
#define LATSZ 64
#define EXPSZ 128
#define HYPSZ 128
#define RNNSZ 256
#define NACTSZ 32
#define SDSZ 128

__device__ __forceinline__ u16 f2bf(float f) {
    unsigned int u = __float_as_uint(f);
    unsigned int r = (u + 0x7FFFu + ((u >> 16) & 1u)) >> 16;
    return (u16)r;
}

__device__ __forceinline__ v16bf mk16(v8us lo, v8us hi) {
    v16us w = __builtin_shufflevector(lo, hi, 0,1,2,3,4,5,6,7,8,9,10,11,12,13,14,15);
    return __builtin_bit_cast(v16bf, w);
}
__device__ __forceinline__ v16bf ld16(const u16* p) {
    v16us w = *(const v16us*)p;
    return __builtin_bit_cast(v16bf, w);
}

// async global -> LDS copy of 16 bytes (per lane); tracked by ASYNCcnt
__device__ __forceinline__ void async_b128(void* lds_dst, const u16* gsrc) {
    lds_u16* lp = (lds_u16*)lds_dst;
    asm volatile("global_load_async_to_lds_b128 %0, %1, off"
                 :: "v"(lp), "v"(gsrc)
                 : "memory");
}
__device__ __forceinline__ void wait_async0() {
    asm volatile("s_wait_asynccnt 0x0" ::: "memory");
}

// ---------------------------------------------------------------- pack f32 -> bf16
__global__ void pack_bf(const float* __restrict__ src, u16* __restrict__ dst, int n) {
    int i = blockIdx.x * blockDim.x + threadIdx.x;
    if (i < n) dst[i] = f2bf(src[i]);
}

// pack hw_b (vector [i*128+o]) transposed to W[o][i] for GEMM form
__global__ void pack_hbT(const float* __restrict__ src, u16* __restrict__ dst) {
    int idx = blockIdx.x * blockDim.x + threadIdx.x;   // 16384
    if (idx < 128 * 128) {
        int o = idx >> 7, ii = idx & 127;
        dst[idx] = f2bf(src[ii * 128 + o]);            // dst[o*128+ii]
    }
}

// ---------------------------------------------------------------- generic WMMA GEMM
// C[M,N] = act( A_bf16[M,K] @ W_bf16[N,K]^T + bias ), optional accumulate into C,
// optional mirrored bf16 output. Block tile 64M x 128N, 8 waves (4Mx2N), K step 32.
__launch_bounds__(256)
__global__ void gemm_bf16_wmma(const u16* __restrict__ A, const u16* __restrict__ W,
                               const float* __restrict__ bias,
                               float* __restrict__ C, u16* __restrict__ Cbf,
                               int M, int N, int K, int act, int accumulate) {
    __shared__ u16 lsA[64][32];
    __shared__ u16 lsW[128][32];
    const int tid  = threadIdx.x;
    const int lane = tid & 31, wid = tid >> 5;
    const int waveM = wid & 3, waveN = wid >> 2;
    const int half = lane >> 4, lrow = lane & 15;
    const int m0 = blockIdx.y * 64;
    const int n0 = blockIdx.x * 128;

    const v8f  vz = {0.f,0.f,0.f,0.f,0.f,0.f,0.f,0.f};
    const v8us uz = {0,0,0,0,0,0,0,0};
    v8f acc[4] = {vz, vz, vz, vz};

    for (int k0 = 0; k0 < K; k0 += 32) {
        __syncthreads();
        {
            int r = tid >> 2;
            int c = (tid & 3) * 8;
            v8us va = uz;
            if (m0 + r < M) va = *(const v8us*)&A[(size_t)(m0 + r) * K + k0 + c];
            *(v8us*)&lsA[r][c] = va;
            v8us w0 = uz, w1 = uz;
            if (n0 + r      < N) w0 = *(const v8us*)&W[(size_t)(n0 + r)      * K + k0 + c];
            if (n0 + r + 64 < N) w1 = *(const v8us*)&W[(size_t)(n0 + r + 64) * K + k0 + c];
            *(v8us*)&lsW[r][c]      = w0;
            *(v8us*)&lsW[r + 64][c] = w1;
        }
        __syncthreads();
        const int rowA = waveM * 16 + lrow;
        v16bf af = mk16(*(const v8us*)&lsA[rowA][half * 16],
                        *(const v8us*)&lsA[rowA][half * 16 + 8]);
#pragma unroll
        for (int nt = 0; nt < 4; ++nt) {
            const int rowW = waveN * 64 + nt * 16 + lrow;
            v16bf bf = ld16(&lsW[rowW][half * 16]);
            acc[nt] = __builtin_amdgcn_wmma_f32_16x16x32_bf16(
                false, af, false, bf, (short)0, acc[nt], false, false);
        }
    }
#pragma unroll
    for (int nt = 0; nt < 4; ++nt) {
#pragma unroll
        for (int r = 0; r < 8; ++r) {
            int row = m0 + waveM * 16 + half * 8 + r;
            int col = n0 + waveN * 64 + nt * 16 + lrow;
            if (row < M && col < N) {
                size_t idx = (size_t)row * N + col;
                float v = acc[nt][r];
                if (bias) v += bias[col];
                if (accumulate) v += C[idx];
                if (act == 1) v = fmaxf(v, 0.f);
                C[idx] = v;
                if (Cbf) Cbf[idx] = f2bf(v);
            }
        }
    }
}

// ---------------------------------------------------------------- batchnorm stats
__global__ void bn_stats(const float* __restrict__ X, float* __restrict__ mean,
                         float* __restrict__ var, int M, int N) {
    __shared__ float s1[256], s2[256];
    int j = blockIdx.x;
    float a = 0.f, b = 0.f;
    for (int i = threadIdx.x; i < M; i += 256) {
        float v = X[(size_t)i * N + j];
        a += v; b += v * v;
    }
    s1[threadIdx.x] = a; s2[threadIdx.x] = b;
    __syncthreads();
    for (int st = 128; st > 0; st >>= 1) {
        if (threadIdx.x < st) {
            s1[threadIdx.x] += s1[threadIdx.x + st];
            s2[threadIdx.x] += s2[threadIdx.x + st];
        }
        __syncthreads();
    }
    if (threadIdx.x == 0) {
        float m = s1[0] / (float)M;
        mean[j] = m;
        var[j] = s2[0] / (float)M - m * m;
    }
}

__global__ void bn_apply_leaky(const float* __restrict__ X, const float* __restrict__ mean,
                               const float* __restrict__ var, const float* __restrict__ g,
                               const float* __restrict__ be,
                               float* __restrict__ Yf, u16* __restrict__ Ybf,
                               long total, int N) {
    long i = (long)blockIdx.x * blockDim.x + threadIdx.x;
    if (i >= total) return;
    int col = (int)(i % N);
    float xn = g[col] * (X[i] - mean[col]) * rsqrtf(var[col] + 1e-5f) + be[col];
    float y = xn > 0.f ? xn : 0.01f * xn;
    if (Yf) Yf[i] = y;
    Ybf[i] = f2bf(y);
}

// ---------------------------------------------------------------- VAE sample
__global__ void vae_sample(const float* __restrict__ lat, const float* __restrict__ noise,
                           float* __restrict__ stdv, u16* __restrict__ latbf) {
    int i = blockIdx.x * blockDim.x + threadIdx.x;   // MB*64
    if (i >= MBSZ * LATSZ) return;
    int b = i >> 6, l = i & 63;
    float mu = lat[b * 128 + l];
    float v  = fmaxf(expf(lat[b * 128 + 64 + l]), 0.002f);
    float sd = sqrtf(v);
    stdv[i] = sd;
    latbf[i] = f2bf(mu + sd * noise[i]);
}

// ---------------------------------------------------------------- fused hypernetwork
// z[b,o] = sum_i x[b,i] * ( lf[b,:] . hw_w[i*128+o, :] ) + extra[b,o]
// Block: 128 batch rows, 8 waves x 16 rows. Per i: 32 KB weight slice.
// LDS: ls[0] first stages latent_final (A-frags hoisted to registers), then
// ls[0]/ls[1] ping-pong the weight slices, filled by ASYNC global->LDS copies
// issued before the WMMA burst and drained with s_wait_asynccnt afterwards.
__launch_bounds__(256)
__global__ void hyper_fused(const u16* __restrict__ lfbf, const float* __restrict__ x,
                            const u16* __restrict__ hww, const float* __restrict__ extra,
                            float* __restrict__ Z) {
    __shared__ u16 ls[2][128][128];   // 2 x 32 KB
    const int tid  = threadIdx.x;
    const int lane = tid & 31, wid = tid >> 5;
    const int half = lane >> 4, lrow = lane & 15;
    const int rowBase = blockIdx.x * 128;

    // stage latent_final block into ls[0] (async, bypasses VGPRs)
    for (int t = tid; t < 128 * 16; t += 256) {
        int r = t >> 4, c8 = (t & 15) * 8;
        async_b128(&ls[0][r][c8], &lfbf[(size_t)(rowBase + r) * 128 + c8]);
    }
    wait_async0();
    __syncthreads();

    const int myRow = wid * 16;
    v16bf afr[4];
#pragma unroll
    for (int kt = 0; kt < 4; ++kt)
        afr[kt] = mk16(*(const v8us*)&ls[0][myRow + lrow][kt * 32 + half * 16],
                       *(const v8us*)&ls[0][myRow + lrow][kt * 32 + half * 16 + 8]);
    __syncthreads();   // all waves done reading lf from ls[0]

    // preload weight slice i=0 into ls[1]
    for (int t = tid; t < 128 * 16; t += 256) {
        int o = t >> 4, c8 = (t & 15) * 8;
        async_b128(&ls[1][o][c8], &hww[(size_t)o * 128 + c8]);
    }
    wait_async0();
    __syncthreads();

    const v8f vz = {0.f,0.f,0.f,0.f,0.f,0.f,0.f,0.f};
    v8f zacc[8];
#pragma unroll
    for (int nt = 0; nt < 8; ++nt) zacc[nt] = vz;

    int cur = 1;
    for (int i = 0; i < 128; ++i) {
        const int nxt = cur ^ 1;
        // kick off async fetch of next slice while we compute on the current one
        if (i + 1 < 128) {
            for (int t = tid; t < 128 * 16; t += 256) {
                int o = t >> 4, c8 = (t & 15) * 8;
                async_b128(&ls[nxt][o][c8],
                           &hww[((size_t)(i + 1) * 128 + o) * 128 + c8]);
            }
        }

        v8f tacc[8];
#pragma unroll
        for (int nt = 0; nt < 8; ++nt) tacc[nt] = vz;
#pragma unroll
        for (int nt = 0; nt < 8; ++nt) {
#pragma unroll
            for (int kt = 0; kt < 4; ++kt) {
                v16bf bfr = ld16(&ls[cur][nt * 16 + lrow][kt * 32 + half * 16]);
                tacc[nt] = __builtin_amdgcn_wmma_f32_16x16x32_bf16(
                    false, afr[kt], false, bfr, (short)0, tacc[nt], false, false);
            }
        }
        float s[8];
#pragma unroll
        for (int r = 0; r < 8; ++r)
            s[r] = x[(size_t)(rowBase + myRow + half * 8 + r) * 128 + i];
#pragma unroll
        for (int nt = 0; nt < 8; ++nt)
#pragma unroll
            for (int r = 0; r < 8; ++r)
                zacc[nt][r] += s[r] * tacc[nt][r];

        wait_async0();     // next slice fully in LDS (this wave's copies)
        __syncthreads();   // ...and everyone else's; also done reading ls[cur]
        cur = nxt;
    }

#pragma unroll
    for (int nt = 0; nt < 8; ++nt)
#pragma unroll
        for (int r = 0; r < 8; ++r) {
            int row = rowBase + myRow + half * 8 + r;
            int col = nt * 16 + lrow;
            size_t idx = (size_t)row * 128 + col;
            Z[idx] = zacc[nt][r] + extra[idx];
        }
}

// ---------------------------------------------------------------- row min-max norm
__global__ void rowminmax_pack(const float* __restrict__ Z, u16* __restrict__ Zbf) {
    __shared__ float smin[128], smax[128];
    int b = blockIdx.x, j = threadIdx.x;
    float v = Z[(size_t)b * 128 + j];
    smin[j] = v; smax[j] = v;
    __syncthreads();
    for (int st = 64; st > 0; st >>= 1) {
        if (j < st) {
            smin[j] = fminf(smin[j], smin[j + st]);
            smax[j] = fmaxf(smax[j], smax[j + st]);
        }
        __syncthreads();
    }
    float zn = (v - smin[0]) / (smax[0] - smin[0] + 1e-12f);
    Zbf[(size_t)b * 128 + j] = f2bf(zn);
}

// ---------------------------------------------------------------- GRU gates
__global__ void gru_gate(const float* __restrict__ gi, const float* __restrict__ gh,
                         const float* __restrict__ hin,
                         float* __restrict__ hout, u16* __restrict__ hbf) {
    int i = blockIdx.x * blockDim.x + threadIdx.x;  // MB*RNN
    if (i >= MBSZ * RNNSZ) return;
    int b = i >> 8, j = i & 255;
    size_t base = (size_t)b * 768;
    float ir = gi[base + j], iz = gi[base + 256 + j], in_ = gi[base + 512 + j];
    float hr = gh[base + j], hz = gh[base + 256 + j], hn  = gh[base + 512 + j];
    float rg = 1.f / (1.f + expf(-(ir + hr)));
    float zg = 1.f / (1.f + expf(-(iz + hz)));
    float ng = tanhf(in_ + rg * hn);
    float h  = (1.f - zg) * ng + zg * hin[i];
    hout[i] = h;
    hbf[i] = f2bf(h);
}

// ---------------------------------------------------------------- inference input concat
__global__ void build_infin(const float* __restrict__ hin, const float* __restrict__ state,
                            u16* __restrict__ dst) {
    int i = blockIdx.x * blockDim.x + threadIdx.x;  // MB*384
    if (i >= MBSZ * 384) return;
    int b = i / 384, j = i - b * 384;
    float v = (j < 256) ? hin[(size_t)b * 256 + j]
                        : state[(size_t)(b >> 3) * 128 + (j - 256)];
    dst[i] = f2bf(v);
}

// ---------------------------------------------------------------- losses
__global__ void loss_init(float* acc) {
    if (blockIdx.x == 0 && threadIdx.x < 8) acc[threadIdx.x] = 0.f;
}

__global__ void loss_entkl(const float* __restrict__ lat, const float* __restrict__ stdv,
                           const float* __restrict__ lat2, float* __restrict__ acc) {
    __shared__ float se[256], sk[256];
    int i = blockIdx.x * 256 + threadIdx.x;  // MB*64
    float e = 0.f, k = 0.f;
    if (i < MBSZ * LATSZ) {
        int b = i >> 6, l = i & 63;
        float sd  = stdv[i];
        float mu  = lat[b * 128 + l];
        float var = sd * sd;
        float mu2  = lat2[b * 128 + l];
        float var2 = fmaxf(expf(lat2[b * 128 + 64 + l]), 0.002f);
        float sd2  = sqrtf(var2);
        e = 1.41893853f + logf(sd);               // 0.5 + 0.5*log(2*pi) + log(std)
        float dm = mu - mu2;
        k = logf(sd2 / sd) + (var + dm * dm) / (2.f * var2) - 0.5f;
    }
    se[threadIdx.x] = e; sk[threadIdx.x] = k;
    __syncthreads();
    for (int st = 128; st > 0; st >>= 1) {
        if (threadIdx.x < st) {
            se[threadIdx.x] += se[threadIdx.x + st];
            sk[threadIdx.x] += sk[threadIdx.x + st];
        }
        __syncthreads();
    }
    if (threadIdx.x == 0) {
        atomicAdd(&acc[0], se[0]);
        atomicAdd(&acc[1], sk[0]);
    }
}

__global__ void loss_cons(const float* __restrict__ lf, float* __restrict__ acc) {
    __shared__ float sc[256];
    int i = blockIdx.x * 256 + threadIdx.x;  // 1024*8*128
    float s = 0.f;
    if (i < 1024 * 8 * 128) {
        int e = i & 127, a = (i >> 7) & 7, g = i >> 10;
        float v = lf[i];
        for (int k = 1; k < 8; ++k) {       // k==8 roll is identity -> zero term
            int a2 = (a - k + 8) & 7;
            float d = v - lf[(size_t)(g * 8 + a2) * 128 + e];
            s += d * d;
        }
    }
    sc[threadIdx.x] = s;
    __syncthreads();
    for (int st = 128; st > 0; st >>= 1) {
        if (threadIdx.x < st) sc[threadIdx.x] += sc[threadIdx.x + st];
        __syncthreads();
    }
    if (threadIdx.x == 0) atomicAdd(&acc[2], sc[0]);
}

__global__ void loss_final(const float* __restrict__ acc, float* __restrict__ out) {
    if (blockIdx.x == 0 && threadIdx.x == 0) {
        float ent  = acc[0] / (float)MBSZ;
        float kl   = acc[1] / (float)MBSZ;
        float cons = acc[2] / (128.f * (float)MBSZ);
        out[0] = 0.001f * ent + 0.0001f * kl + 0.001f * cons;
    }
}

// ---------------------------------------------------------------- host side
static inline size_t al256(size_t x) { return (x + 255) & ~(size_t)255; }

static void launch_gemm(const u16* A, const u16* W, const float* bias, float* C, u16* Cbf,
                        int M, int N, int K, int act, int accum, hipStream_t s) {
    dim3 g((N + 127) / 128, (M + 63) / 64);
    gemm_bf16_wmma<<<g, 256, 0, s>>>(A, W, bias, C, Cbf, M, N, K, act, accum);
}

static void launch_pack(const float* src, u16* dst, int n, hipStream_t s) {
    pack_bf<<<(n + 255) / 256, 256, 0, s>>>(src, dst, n);
}

extern "C" void kernel_launch(void* const* d_in, const int* in_sizes, int n_in,
                              void* d_out, int out_size, void* d_ws, size_t ws_size,
                              hipStream_t stream) {
    (void)in_sizes; (void)n_in; (void)out_size; (void)ws_size;
    const float* inputs   = (const float*)d_in[0];
    const float* hidden   = (const float*)d_in[1];
    const float* state    = (const float*)d_in[2];
    const float* noise    = (const float*)d_in[3];
    const float* e_w1 = (const float*)d_in[4];  const float* e_b1 = (const float*)d_in[5];
    const float* e_g1 = (const float*)d_in[6];  const float* e_be1= (const float*)d_in[7];
    const float* e_w2 = (const float*)d_in[8];  const float* e_b2 = (const float*)d_in[9];
    const float* i_w1 = (const float*)d_in[10]; const float* i_b1 = (const float*)d_in[11];
    const float* i_g1 = (const float*)d_in[12]; const float* i_be1= (const float*)d_in[13];
    const float* i_w2 = (const float*)d_in[14]; const float* i_b2 = (const float*)d_in[15];
    const float* l_w  = (const float*)d_in[16]; const float* l_b  = (const float*)d_in[17];
    const float* l_g  = (const float*)d_in[18]; const float* l_be = (const float*)d_in[19];
    const float* hw_w = (const float*)d_in[20]; const float* hw_b = (const float*)d_in[21];
    const float* hb_w = (const float*)d_in[22]; const float* hb_b = (const float*)d_in[23];
    const float* fc1_w= (const float*)d_in[24]; const float* fc1_b= (const float*)d_in[25];
    const float* wih  = (const float*)d_in[26]; const float* whh  = (const float*)d_in[27];
    const float* bih  = (const float*)d_in[28]; const float* bhh  = (const float*)d_in[29];
    const float* fc2_w= (const float*)d_in[30]; const float* fc2_b= (const float*)d_in[31];

    float* out    = (float*)d_out;
    float* q_out  = out;                          // [8192,32]
    float* h_out  = out + MBSZ * NACTSZ;          // [8192,256]
    float* l_out  = out + MBSZ * NACTSZ + MBSZ * RNNSZ;  // scalar

    // ---- workspace bump allocator
    char* w = (char*)d_ws;
    size_t off = 0;
#define WALLOC(T, name, elems) T* name = (T*)(w + off); off += al256((size_t)(elems) * sizeof(T))
    WALLOC(u16,   in_bf,    MBSZ * INSZ);
    WALLOC(u16,   ew1_bf,   GHSZ * INSZ);
    WALLOC(u16,   ew2_bf,   128 * GHSZ);
    WALLOC(u16,   iw1_bf,   GHSZ * 384);
    WALLOC(u16,   iw2_bf,   128 * GHSZ);
    WALLOC(u16,   lw_bf,    EXPSZ * LATSZ);
    WALLOC(u16,   hww_bf,   INSZ * HYPSZ * EXPSZ);
    WALLOC(u16,   hbw_bf,   HYPSZ * EXPSZ);
    WALLOC(u16,   hbT_bf,   HYPSZ * INSZ);
    WALLOC(u16,   fc1w_bf,  RNNSZ * HYPSZ);
    WALLOC(u16,   wih_bf,   3 * RNNSZ * RNNSZ);
    WALLOC(u16,   whh_bf,   3 * RNNSZ * RNNSZ);
    WALLOC(u16,   fc2w_bf,  NACTSZ * RNNSZ);
    WALLOC(u16,   hin_bf,   MBSZ * RNNSZ);
    WALLOC(u16,   infin_bf, MBSZ * 384);
    WALLOC(float, big,      MBSZ * GHSZ);        // enc1 pre-bn, reused for inf pre-bn
    WALLOC(u16,   bigact_bf,MBSZ * GHSZ);        // enc1 activ bf, reused for inf
    WALLOC(float, meanb,    GHSZ);
    WALLOC(float, varb,     GHSZ);
    WALLOC(float, lat,      MBSZ * 128);
    WALLOC(float, stdv,     MBSZ * LATSZ);
    WALLOC(u16,   lat_bf,   MBSZ * LATSZ);
    WALLOC(float, lpre,     MBSZ * EXPSZ);
    WALLOC(float, lf_f32,   MBSZ * EXPSZ);
    WALLOC(u16,   lf_bf,    MBSZ * EXPSZ);
    WALLOC(float, extra,    MBSZ * HYPSZ);
    WALLOC(float, zbuf,     MBSZ * HYPSZ);
    WALLOC(u16,   z_bf,     MBSZ * HYPSZ);
    WALLOC(float, xbuf,     MBSZ * RNNSZ);
    WALLOC(u16,   x_bf,     MBSZ * RNNSZ);
    WALLOC(float, gi,       MBSZ * 3 * RNNSZ);
    WALLOC(float, gh,       MBSZ * 3 * RNNSZ);
    WALLOC(u16,   h_bf,     MBSZ * RNNSZ);
    WALLOC(float, lat2,     MBSZ * 128);
    WALLOC(float, acc,      8);
#undef WALLOC

    // ---- pack inputs & weights to bf16
    launch_pack(inputs, in_bf,   MBSZ * INSZ, stream);
    launch_pack(hidden, hin_bf,  MBSZ * RNNSZ, stream);
    launch_pack(e_w1,  ew1_bf,   GHSZ * INSZ, stream);
    launch_pack(e_w2,  ew2_bf,   128 * GHSZ, stream);
    launch_pack(i_w1,  iw1_bf,   GHSZ * 384, stream);
    launch_pack(i_w2,  iw2_bf,   128 * GHSZ, stream);
    launch_pack(l_w,   lw_bf,    EXPSZ * LATSZ, stream);
    launch_pack(hw_w,  hww_bf,   INSZ * HYPSZ * EXPSZ, stream);
    launch_pack(hb_w,  hbw_bf,   HYPSZ * EXPSZ, stream);
    launch_pack(fc1_w, fc1w_bf,  RNNSZ * HYPSZ, stream);
    launch_pack(wih,   wih_bf,   3 * RNNSZ * RNNSZ, stream);
    launch_pack(whh,   whh_bf,   3 * RNNSZ * RNNSZ, stream);
    launch_pack(fc2_w, fc2w_bf,  NACTSZ * RNNSZ, stream);
    pack_hbT<<<(128 * 128 + 255) / 256, 256, 0, stream>>>(hw_b, hbT_bf);

    // ---- encoder: lin -> bn -> leaky -> lin
    launch_gemm(in_bf, ew1_bf, e_b1, big, nullptr, MBSZ, GHSZ, INSZ, 0, 0, stream);
    bn_stats<<<GHSZ, 256, 0, stream>>>(big, meanb, varb, MBSZ, GHSZ);
    bn_apply_leaky<<<(MBSZ * GHSZ + 255) / 256, 256, 0, stream>>>(
        big, meanb, varb, e_g1, e_be1, nullptr, bigact_bf, (long)MBSZ * GHSZ, GHSZ);
    launch_gemm(bigact_bf, ew2_bf, e_b2, lat, nullptr, MBSZ, 128, GHSZ, 0, 0, stream);

    // ---- VAE sample -> latent path
    vae_sample<<<(MBSZ * LATSZ + 255) / 256, 256, 0, stream>>>(lat, noise, stdv, lat_bf);
    launch_gemm(lat_bf, lw_bf, l_b, lpre, nullptr, MBSZ, EXPSZ, LATSZ, 0, 0, stream);
    bn_stats<<<EXPSZ, 256, 0, stream>>>(lpre, meanb, varb, MBSZ, EXPSZ);
    bn_apply_leaky<<<(MBSZ * EXPSZ + 255) / 256, 256, 0, stream>>>(
        lpre, meanb, varb, l_g, l_be, lf_f32, lf_bf, (long)MBSZ * EXPSZ, EXPSZ);

    // ---- hypernetwork: extra = ib + x@hw_b_reshaped ; fused bilinear z
    launch_gemm(lf_bf, hbw_bf, hb_b, extra, nullptr, MBSZ, HYPSZ, EXPSZ, 0, 0, stream);
    launch_gemm(in_bf, hbT_bf, nullptr, extra, nullptr, MBSZ, HYPSZ, INSZ, 0, 1, stream);
    hyper_fused<<<MBSZ / 128, 256, 0, stream>>>(lf_bf, inputs, hww_bf, extra, zbuf);
    rowminmax_pack<<<MBSZ, 128, 0, stream>>>(zbuf, z_bf);

    // ---- fc1 + GRU + q
    launch_gemm(z_bf, fc1w_bf, fc1_b, xbuf, x_bf, MBSZ, RNNSZ, HYPSZ, 1, 0, stream);
    launch_gemm(x_bf,  wih_bf, bih, gi, nullptr, MBSZ, 3 * RNNSZ, RNNSZ, 0, 0, stream);
    launch_gemm(hin_bf, whh_bf, bhh, gh, nullptr, MBSZ, 3 * RNNSZ, RNNSZ, 0, 0, stream);
    gru_gate<<<(MBSZ * RNNSZ + 255) / 256, 256, 0, stream>>>(gi, gh, hidden, h_out, h_bf);
    launch_gemm(h_bf, fc2w_bf, fc2_b, q_out, nullptr, MBSZ, NACTSZ, RNNSZ, 0, 0, stream);

    // ---- inference encoder
    build_infin<<<(MBSZ * 384 + 255) / 256, 256, 0, stream>>>(hidden, state, infin_bf);
    launch_gemm(infin_bf, iw1_bf, i_b1, big, nullptr, MBSZ, GHSZ, 384, 0, 0, stream);
    bn_stats<<<GHSZ, 256, 0, stream>>>(big, meanb, varb, MBSZ, GHSZ);
    bn_apply_leaky<<<(MBSZ * GHSZ + 255) / 256, 256, 0, stream>>>(
        big, meanb, varb, i_g1, i_be1, nullptr, bigact_bf, (long)MBSZ * GHSZ, GHSZ);
    launch_gemm(bigact_bf, iw2_bf, i_b2, lat2, nullptr, MBSZ, 128, GHSZ, 0, 0, stream);

    // ---- losses
    loss_init<<<1, 32, 0, stream>>>(acc);
    loss_entkl<<<(MBSZ * LATSZ + 255) / 256, 256, 0, stream>>>(lat, stdv, lat2, acc);
    loss_cons<<<(1024 * 8 * 128 + 255) / 256, 256, 0, stream>>>(lf_f32, acc);
    loss_final<<<1, 32, 0, stream>>>(acc, l_out);
}
